// EfficientAttention_6631429505139
// MI455X (gfx1250) — compile-verified, matrix operand order checked by reading, not run on hardware
//
#include <hip/hip_runtime.h>
#include <cstdint>

#define DIM     2048
#define NHEADS  16
#define HDIM    128
#define SEQ     2048
#define BATCH   2

typedef __attribute__((ext_vector_type(16))) __bf16 v16bf;
typedef __attribute__((ext_vector_type(8)))  float  v8f;

union Frag {
    v16bf    v;
    uint4    u[2];
    uint16_t h[16];
};

__device__ __forceinline__ v8f zero8() {
    v8f z = {0.f, 0.f, 0.f, 0.f, 0.f, 0.f, 0.f, 0.f};
    return z;
}

// hardware f32 -> bf16 conversion (v_cvt on gfx1250)
__device__ __forceinline__ uint16_t f2bf(float f) {
    union { __bf16 b; uint16_t u; } c;
    c.b = (__bf16)f;
    return c.u;
}

__device__ __forceinline__ uint32_t pack2(float a, float b) {
    return (uint32_t)f2bf(a) | ((uint32_t)f2bf(b) << 16);
}

// CDNA5 async global->LDS copy (ASYNCcnt); IOFFSET applies to both addresses.
__device__ __forceinline__ void async_copy32(uint32_t lds_addr, const void* gptr) {
    asm volatile(
        "global_load_async_to_lds_b128 %0, %1, off\n\t"
        "global_load_async_to_lds_b128 %0, %1, off offset:16"
        :: "v"(lds_addr), "v"(gptr) : "memory");
}
__device__ __forceinline__ void wait_async0() {
    asm volatile("s_wait_asynccnt 0" ::: "memory");
}

// ---------------------------------------------------------------------------
// fp32 -> bf16 conversion (4 elements / thread)
// ---------------------------------------------------------------------------
__global__ __launch_bounds__(256)
void k_f32_to_bf16(const float* __restrict__ s, uint16_t* __restrict__ d, int n) {
    int i = (blockIdx.x * blockDim.x + threadIdx.x) * 4;
    if (i >= n) return;
    float4 f = *(const float4*)(s + i);
    uint2 o;
    o.x = pack2(f.x, f.y);
    o.y = pack2(f.z, f.w);
    *(uint2*)(d + i) = o;
}

// ---------------------------------------------------------------------------
// C[M,N] = A[M,K](bf16) * Bw[N,K]^T (bf16)  -- i.e. y = x @ W.T
// 128x128 block tile, 256 threads = 8 waves (2x4), wave tile 64x32.
// LDS tiles staged with GLOBAL_LOAD_ASYNC_TO_LDS_B128.
// OUT_BF16 ? bf16 output : fp32 output
// ---------------------------------------------------------------------------
template <int OUT_BF16>
__global__ __launch_bounds__(256)
void gemm_bf16(const uint16_t* __restrict__ A,
               const uint16_t* __restrict__ Bw,
               void* __restrict__ Cout,
               int M, int N, int K) {
    __shared__ uint16_t Asl[128 * 32];
    __shared__ uint16_t Bsl[128 * 32];

    const int tid  = threadIdx.x;
    const int lane = tid & 31;
    const int wv   = tid >> 5;
    const int wm   = wv & 1;        // 2 waves along M
    const int wn   = wv >> 1;       // 4 waves along N
    const int bm   = blockIdx.y * 128;
    const int bn   = blockIdx.x * 128;

    v8f acc[4][2];
#pragma unroll
    for (int i = 0; i < 4; ++i)
#pragma unroll
        for (int j = 0; j < 2; ++j) acc[i][j] = zero8();

    // cooperative LDS fill: 2 threads per 32-element row, 32 B each
    const int lrow  = tid >> 1;          // 0..127
    const int lhalf = (tid & 1) * 16;    // 0 or 16
    const uint16_t* Ag = A  + (size_t)(bm + lrow) * K + lhalf;
    const uint16_t* Bg = Bw + (size_t)(bn + lrow) * K + lhalf;
    const uint32_t a_lds = (uint32_t)(uintptr_t)&Asl[lrow * 32 + lhalf];
    const uint32_t b_lds = (uint32_t)(uintptr_t)&Bsl[lrow * 32 + lhalf];

    const int fr = lane & 15;            // fragment row within 16
    const int fc = (lane >> 4) * 16;     // K half select

    for (int k0 = 0; k0 < K; k0 += 32) {
        __syncthreads();                 // previous iter done reading LDS
        async_copy32(a_lds, Ag + k0);
        async_copy32(b_lds, Bg + k0);
        if (k0 + 32 < K) {
            __builtin_prefetch(Ag + k0 + 32, 0, 0);   // global_prefetch_b8
            __builtin_prefetch(Bg + k0 + 32, 0, 0);
        }
        wait_async0();                   // own async writes landed in LDS
        __syncthreads();                 // everyone's writes visible

        Frag af[4], bf2[2];
#pragma unroll
        for (int i = 0; i < 4; ++i) {
            const uint16_t* p = &Asl[(wm * 64 + i * 16 + fr) * 32 + fc];
            af[i].u[0] = ((const uint4*)p)[0];
            af[i].u[1] = ((const uint4*)p)[1];
        }
#pragma unroll
        for (int j = 0; j < 2; ++j) {
            const uint16_t* p = &Bsl[(wn * 32 + j * 16 + fr) * 32 + fc];
            bf2[j].u[0] = ((const uint4*)p)[0];
            bf2[j].u[1] = ((const uint4*)p)[1];
        }
#pragma unroll
        for (int i = 0; i < 4; ++i)
#pragma unroll
            for (int j = 0; j < 2; ++j)
                acc[i][j] = __builtin_amdgcn_wmma_f32_16x16x32_bf16(
                    false, af[i].v, false, bf2[j].v, (short)0, acc[i][j],
                    false, false);
    }

    // D layout: lanes 0-15 VGPRr -> (M=r, N=lane); lanes 16-31 -> (M=8+r, N=lane-16)
    const int cn = lane & 15;
    const int rr = (lane >> 4) * 8;
#pragma unroll
    for (int i = 0; i < 4; ++i)
#pragma unroll
        for (int j = 0; j < 2; ++j) {
            int col = bn + wn * 32 + j * 16 + cn;
#pragma unroll
            for (int r = 0; r < 8; ++r) {
                int row = bm + wm * 64 + i * 16 + rr + r;
                float vres = acc[i][j][r];
                if (OUT_BF16)
                    ((uint16_t*)Cout)[(size_t)row * N + col] = f2bf(vres);
                else
                    ((float*)Cout)[(size_t)row * N + col] = vres;
            }
        }
}

// ---------------------------------------------------------------------------
// Causal flash attention. Q,K,V bf16 laid out as [B*S, DIM] with head h at
// column h*HDIM. One block per (b, h, 128 query rows); 8 waves x 16 q-rows.
// ---------------------------------------------------------------------------
__global__ __launch_bounds__(256)
void attn_kernel(const uint16_t* __restrict__ Q,
                 const uint16_t* __restrict__ K,
                 const uint16_t* __restrict__ V,
                 uint16_t* __restrict__ O) {
    __shared__ uint16_t Ks[32 * 128];      // [kv][d]
    __shared__ uint16_t Vts[128 * 32];     // transposed: [d][kv]
    __shared__ uint16_t Ps[8 * 32 * 16];   // per-wave P, column-major [kv][q]

    const int tid  = threadIdx.x;
    const int lane = tid & 31;
    const int wv   = tid >> 5;
    const int h    = blockIdx.y & (NHEADS - 1);
    const int b    = blockIdx.y >> 4;
    const int qb   = blockIdx.x * 128;
    const int qw   = qb + wv * 16;
    const size_t headoff = (size_t)h * HDIM;

    const int fr = lane & 15;
    const int fc = (lane >> 4) * 16;

    // preload this wave's Q fragments: 16 rows x 128 dims = 4 K-chunks of 32
    Frag qf[4];
    {
        const uint16_t* qp =
            Q + (size_t)(b * SEQ + qw + fr) * DIM + headoff + fc;
#pragma unroll
        for (int c = 0; c < 4; ++c) {
            qf[c].u[0] = ((const uint4*)(qp + c * 32))[0];
            qf[c].u[1] = ((const uint4*)(qp + c * 32))[1];
        }
    }

    const float NEG_INF = -__builtin_inff();
    float mrow[8], lrow[8];
#pragma unroll
    for (int r = 0; r < 8; ++r) { mrow[r] = NEG_INF; lrow[r] = 0.f; }
    v8f oacc[8];
#pragma unroll
    for (int n = 0; n < 8; ++n) oacc[n] = zero8();

    const int ldrow = tid >> 3;          // kv row 0..31
    const int ldseg = (tid & 7) * 16;    // d offset
    const float sc = 0.08838834764831845f * 1.4426950408889634f; // 1/sqrt(128)*log2e
    const int qmaxw = qw + 15;

    for (int kv0 = 0; kv0 < qb + 128; kv0 += 32) {
        __syncthreads();
        {
            const uint16_t* kp =
                K + (size_t)(b * SEQ + kv0 + ldrow) * DIM + headoff + ldseg;
            uint16_t* ks = &Ks[ldrow * 128 + ldseg];
            ((uint4*)ks)[0] = ((const uint4*)kp)[0];
            ((uint4*)ks)[1] = ((const uint4*)kp)[1];

            const uint16_t* vp =
                V + (size_t)(b * SEQ + kv0 + ldrow) * DIM + headoff + ldseg;
            Frag vf;
            vf.u[0] = ((const uint4*)vp)[0];
            vf.u[1] = ((const uint4*)vp)[1];
#pragma unroll
            for (int i = 0; i < 16; ++i)
                Vts[(ldseg + i) * 32 + ldrow] = vf.h[i];   // transpose into LDS
        }
        __syncthreads();

        if (kv0 > qmaxw) continue;  // wave fully above diagonal: loads only

        // scores: two 16-wide kv tiles, accumulated over 4 chunks of 32 dims
        v8f s[2];
        s[0] = zero8();
        s[1] = zero8();
#pragma unroll
        for (int t = 0; t < 2; ++t)
#pragma unroll
            for (int c = 0; c < 4; ++c) {
                Frag kf;
                const uint16_t* p = &Ks[(t * 16 + fr) * 128 + c * 32 + fc];
                kf.u[0] = ((const uint4*)p)[0];
                kf.u[1] = ((const uint4*)p)[1];
                s[t] = __builtin_amdgcn_wmma_f32_16x16x32_bf16(
                    false, qf[c].v, false, kf.v, (short)0, s[t], false, false);
            }

        // scale (pre-multiplied by log2e for exp2) + causal mask
        const int qoff = qw + ((lane >> 4) << 3);   // q row for r=0
        const int kvl  = kv0 + (lane & 15);
#pragma unroll
        for (int t = 0; t < 2; ++t) {
            int kva = kvl + 16 * t;
#pragma unroll
            for (int r = 0; r < 8; ++r) {
                float x = s[t][r] * sc;
                s[t][r] = (kva <= qoff + r) ? x : NEG_INF;
            }
        }

        // online softmax: row reductions across the 16-lane half
        float alpha[8];
#pragma unroll
        for (int r = 0; r < 8; ++r) {
            float tmax = fmaxf(s[0][r], s[1][r]);
            tmax = fmaxf(tmax, __shfl_xor(tmax, 1));
            tmax = fmaxf(tmax, __shfl_xor(tmax, 2));
            tmax = fmaxf(tmax, __shfl_xor(tmax, 4));
            tmax = fmaxf(tmax, __shfl_xor(tmax, 8));
            float mnew = fmaxf(mrow[r], tmax);
            alpha[r] = __builtin_amdgcn_exp2f(mrow[r] - mnew);
            float p0 = __builtin_amdgcn_exp2f(s[0][r] - mnew);
            float p1 = __builtin_amdgcn_exp2f(s[1][r] - mnew);
            s[0][r] = p0;
            s[1][r] = p1;
            float ps = p0 + p1;
            ps += __shfl_xor(ps, 1);
            ps += __shfl_xor(ps, 2);
            ps += __shfl_xor(ps, 4);
            ps += __shfl_xor(ps, 8);
            lrow[r] = lrow[r] * alpha[r] + ps;
            mrow[r] = mnew;
        }
#pragma unroll
        for (int n = 0; n < 8; ++n)
#pragma unroll
            for (int r = 0; r < 8; ++r) oacc[n][r] *= alpha[r];

        // stage P (bf16) into per-wave LDS column-major [kv][q], then read
        // back in A-matrix layout (row = q, 16 contiguous kv per lane).
        uint16_t* pw = &Ps[wv * 512];
#pragma unroll
        for (int t = 0; t < 2; ++t) {
            uint4 pk;
            pk.x = pack2(s[t][0], s[t][1]);
            pk.y = pack2(s[t][2], s[t][3]);
            pk.z = pack2(s[t][4], s[t][5]);
            pk.w = pack2(s[t][6], s[t][7]);
            *(uint4*)&pw[(t * 16 + (lane & 15)) * 16 + ((lane >> 4) * 8)] = pk;
        }
        Frag pf;
#pragma unroll
        for (int j = 0; j < 16; ++j)
            pf.h[j] = pw[(fc + j) * 16 + fr];

        // O += P(16x32) * V(32x16 per d-chunk); Vts rows are d, contiguous kv
#pragma unroll
        for (int n = 0; n < 8; ++n) {
            Frag vfr;
            const uint16_t* p = &Vts[(n * 16 + fr) * 32 + fc];
            vfr.u[0] = ((const uint4*)p)[0];
            vfr.u[1] = ((const uint4*)p)[1];
            oacc[n] = __builtin_amdgcn_wmma_f32_16x16x32_bf16(
                false, pf.v, false, vfr.v, (short)0, oacc[n], false, false);
        }
    }

    // normalize and store bf16 at [b, q, h, d]
    const int cn = lane & 15;
    const int rr = (lane >> 4) * 8;
    float inv[8];
#pragma unroll
    for (int r = 0; r < 8; ++r) inv[r] = 1.0f / lrow[r];
#pragma unroll
    for (int n = 0; n < 8; ++n)
#pragma unroll
        for (int r = 0; r < 8; ++r) {
            int q = qw + rr + r;
            int d = n * 16 + cn;
            O[(size_t)(b * SEQ + q) * DIM + headoff + d] =
                f2bf(oacc[n][r] * inv[r]);
        }
}

// ---------------------------------------------------------------------------
// Host launcher
// ---------------------------------------------------------------------------
extern "C" void kernel_launch(void* const* d_in, const int* in_sizes, int n_in,
                              void* d_out, int out_size, void* d_ws,
                              size_t ws_size, hipStream_t stream) {
    (void)in_sizes; (void)n_in; (void)out_size; (void)ws_size;

    const float* X  = (const float*)d_in[0];
    const float* Wq = (const float*)d_in[1];
    const float* Wk = (const float*)d_in[2];
    const float* Wv = (const float*)d_in[3];
    const float* Wo = (const float*)d_in[4];

    const int NX = BATCH * SEQ * DIM;  // 8,388,608
    const int NW = DIM * DIM;          // 4,194,304
    const size_t XB_BYTES = (size_t)NX * 2;  // 16 MB
    const size_t WB_BYTES = (size_t)NW * 2;  // 8 MB

    char* ws = (char*)d_ws;
    uint16_t* Xb  = (uint16_t*)(ws);
    uint16_t* Wqb = (uint16_t*)(ws + XB_BYTES);
    uint16_t* Wkb = (uint16_t*)(ws + XB_BYTES + WB_BYTES);
    uint16_t* Wvb = (uint16_t*)(ws + XB_BYTES + 2 * WB_BYTES);
    uint16_t* Wob = (uint16_t*)(ws + XB_BYTES + 3 * WB_BYTES);
    char* base2 = ws + XB_BYTES + 4 * WB_BYTES;
    uint16_t* Qb = (uint16_t*)(base2);
    uint16_t* Kb = (uint16_t*)(base2 + XB_BYTES);
    uint16_t* Vb = (uint16_t*)(base2 + 2 * XB_BYTES);
    uint16_t* Ab = (uint16_t*)(base2 + 3 * XB_BYTES);
    // total workspace need: 16 + 32 + 48 + 16 = 112 MB

    k_f32_to_bf16<<<NX / 1024, 256, 0, stream>>>(X,  Xb,  NX);
    k_f32_to_bf16<<<NW / 1024, 256, 0, stream>>>(Wq, Wqb, NW);
    k_f32_to_bf16<<<NW / 1024, 256, 0, stream>>>(Wk, Wkb, NW);
    k_f32_to_bf16<<<NW / 1024, 256, 0, stream>>>(Wv, Wvb, NW);
    k_f32_to_bf16<<<NW / 1024, 256, 0, stream>>>(Wo, Wob, NW);

    dim3 gg(DIM / 128, (BATCH * SEQ) / 128);   // (16, 32)
    gemm_bf16<1><<<gg, 256, 0, stream>>>(Xb, Wqb, Qb, BATCH * SEQ, DIM, DIM);
    gemm_bf16<1><<<gg, 256, 0, stream>>>(Xb, Wkb, Kb, BATCH * SEQ, DIM, DIM);
    gemm_bf16<1><<<gg, 256, 0, stream>>>(Xb, Wvb, Vb, BATCH * SEQ, DIM, DIM);

    attn_kernel<<<dim3(SEQ / 128, BATCH * NHEADS), 256, 0, stream>>>(Qb, Kb, Vb, Ab);

    gemm_bf16<0><<<gg, 256, 0, stream>>>(Ab, Wob, d_out, BATCH * SEQ, DIM, DIM);
}
